// SUPEncoder_53068615910239
// MI455X (gfx1250) — matrix operand order repaired and checked by reading, not compiled, not run on hardware
//
#include <hip/hip_runtime.h>
#include <hip/hip_bf16.h>
#include <math.h>

// ---- problem constants (match reference) ----
#define NGRAPH 64
#define DIM    64
#define NFEAT  64
#define NNODES 5000
#define NEDGES 20000
#define EFEAT  5
#define HID    128
#define DD     (DIM*DIM)          // 4096

typedef __attribute__((ext_vector_type(16))) _Float16 v16h;
typedef __attribute__((ext_vector_type(8)))  _Float16 v8h;
typedef __attribute__((ext_vector_type(8)))  float    v8f;

__device__ __forceinline__ float sigmoidf_(float x) { return 1.0f / (1.0f + expf(-x)); }

// ---------------- utility ----------------
__global__ void zero_f32(float* p, int n) {
    int t = blockIdx.x * blockDim.x + threadIdx.x;
    if (t < n) p[t] = 0.0f;
}

// out = relu(x @ lin0_w + lin0_b)   [N,64]
__global__ void lin0_kernel(const float* __restrict__ x, const float* __restrict__ w,
                            const float* __restrict__ b, float* __restrict__ h) {
    int t = blockIdx.x * blockDim.x + threadIdx.x;
    if (t >= NNODES * DIM) return;
    int n = t >> 6, j = t & 63;
    const float* xv = x + n * NFEAT;
    float acc = b[j];
    #pragma unroll 8
    for (int k = 0; k < NFEAT; ++k) acc += xv[k] * w[k * DIM + j];
    h[t] = fmaxf(acc, 0.0f);
}

// hid = relu(edge_attr @ nn_w1 + nn_b1) -> f16  [E,128]
__global__ void edge_mlp1_kernel(const float* __restrict__ ea, const float* __restrict__ w1,
                                 const float* __restrict__ b1, _Float16* __restrict__ hid) {
    int t = blockIdx.x * blockDim.x + threadIdx.x;
    if (t >= NEDGES * HID) return;
    int e = t >> 7, c = t & 127;
    const float* ev = ea + e * EFEAT;
    float acc = b1[c];
    #pragma unroll
    for (int k = 0; k < EFEAT; ++k) acc += ev[k] * w1[k * HID + c];
    hid[t] = (_Float16)fmaxf(acc, 0.0f);
}

// w2t[n][k] = (f16) nn_w2[k][n]   [4096,128]
__global__ void w2_transpose_kernel(const float* __restrict__ w2, _Float16* __restrict__ w2t) {
    int t = blockIdx.x * blockDim.x + threadIdx.x;
    if (t >= DD * HID) return;
    int n = t >> 7, k = t & 127;
    w2t[t] = (_Float16)w2[k * DD + n];
}

// ew[E,4096] (f16) = hid[E,128] @ w2[128,4096] + b2 ; WMMA f32_16x16x32_f16
// one wave computes a 16x64 output block (1 A-fragment reused by 4 B-fragments).
// supertiles = (20000/16) x (4096/64) = 1250 x 64 = 80000 waves, 8 waves/block.
__global__ void ew_gemm_wmma_kernel(const _Float16* __restrict__ hid,
                                    const _Float16* __restrict__ w2t,
                                    const float* __restrict__ b2,
                                    _Float16* __restrict__ ew) {
    const int wave = threadIdx.x >> 5;
    const int lane = threadIdx.x & 31;
    const int tile = blockIdx.x * 8 + wave;          // < 80000 always
    const int tM = tile >> 6;                        // 0..1249
    const int tSN = tile & 63;                       // 0..63  (64-wide col block)
    const int g  = lane >> 4;                        // lane group
    const int mr = lane & 15;

    const _Float16* ap = hid + (size_t)(tM * 16 + mr) * HID;          // A row (edge)
    const _Float16* bp0 = w2t + (size_t)(tSN * 64 + mr) * HID;        // B col, n-tile 0
    const _Float16* bp1 = bp0 + (size_t)16 * HID;
    const _Float16* bp2 = bp0 + (size_t)32 * HID;
    const _Float16* bp3 = bp0 + (size_t)48 * HID;

    v8f acc0 = {}, acc1 = {}, acc2 = {}, acc3 = {};
    #pragma unroll
    for (int kk = 0; kk < HID; kk += 32) {
        v8h a0 = *(const v8h*)(ap + kk + g * 16);
        v8h a1 = *(const v8h*)(ap + kk + g * 16 + 8);
        v16h A;
        #pragma unroll
        for (int j = 0; j < 8; ++j) { A[j] = a0[j]; A[j + 8] = a1[j]; }

        v8h b0, b1;
        v16h B;
        b0 = *(const v8h*)(bp0 + kk + g * 16);
        b1 = *(const v8h*)(bp0 + kk + g * 16 + 8);
        #pragma unroll
        for (int j = 0; j < 8; ++j) { B[j] = b0[j]; B[j + 8] = b1[j]; }
        acc0 = __builtin_amdgcn_wmma_f32_16x16x32_f16(false, A, false, B, (short)0, acc0, false, false);

        b0 = *(const v8h*)(bp1 + kk + g * 16);
        b1 = *(const v8h*)(bp1 + kk + g * 16 + 8);
        #pragma unroll
        for (int j = 0; j < 8; ++j) { B[j] = b0[j]; B[j + 8] = b1[j]; }
        acc1 = __builtin_amdgcn_wmma_f32_16x16x32_f16(false, A, false, B, (short)0, acc1, false, false);

        b0 = *(const v8h*)(bp2 + kk + g * 16);
        b1 = *(const v8h*)(bp2 + kk + g * 16 + 8);
        #pragma unroll
        for (int j = 0; j < 8; ++j) { B[j] = b0[j]; B[j + 8] = b1[j]; }
        acc2 = __builtin_amdgcn_wmma_f32_16x16x32_f16(false, A, false, B, (short)0, acc2, false, false);

        b0 = *(const v8h*)(bp3 + kk + g * 16);
        b1 = *(const v8h*)(bp3 + kk + g * 16 + 8);
        #pragma unroll
        for (int j = 0; j < 8; ++j) { B[j] = b0[j]; B[j + 8] = b1[j]; }
        acc3 = __builtin_amdgcn_wmma_f32_16x16x32_f16(false, A, false, B, (short)0, acc3, false, false);
    }

    // epilogue: D layout -> row = tM*16 + r + 8*g, col = colbase + mr
    const int rowb = tM * 16 + 8 * g;
    #pragma unroll
    for (int nt = 0; nt < 4; ++nt) {
        const v8f acc = (nt == 0) ? acc0 : (nt == 1) ? acc1 : (nt == 2) ? acc2 : acc3;
        const int ccol = tSN * 64 + nt * 16 + mr;
        const float bias = b2[ccol];
        #pragma unroll
        for (int r = 0; r < 8; ++r)
            ew[(size_t)(rowb + r) * DD + ccol] = (_Float16)(acc[r] + bias);
    }
}

// deg[dst] += 1
__global__ void deg_kernel(const int* __restrict__ dst, float* __restrict__ deg) {
    int e = blockIdx.x * blockDim.x + threadIdx.x;
    if (e < NEDGES) atomicAdd(&deg[dst[e]], 1.0f);
}

// msg = out[src] @ ew[e]; scatter-add into agg. one wave per edge.
// lane covers columns 8*(lane&7)..+7 for rows == (lane>>3) mod 4; b128 loads,
// then shfl_xor reduction over the 4 row-subgroups.
__global__ void msg_scatter_kernel(const float* __restrict__ h, const _Float16* __restrict__ ew,
                                   const int* __restrict__ src, const int* __restrict__ dst,
                                   float* __restrict__ agg) {
    const int wave = threadIdx.x >> 5;
    const int lane = threadIdx.x & 31;
    const int e = blockIdx.x * 8 + wave;             // 2500*8 = 20000 exactly
    const int s = src[e], d = dst[e];
    const float* xv = h + s * DIM;
    const _Float16* wp = ew + (size_t)e * DD;
    const int g  = lane >> 3;        // row subgroup 0..3
    const int c0 = (lane & 7) * 8;   // column base

    float acc[8] = {0.f, 0.f, 0.f, 0.f, 0.f, 0.f, 0.f, 0.f};
    #pragma unroll
    for (int i = 0; i < DIM; i += 4) {
        const int row = i + g;
        const float xi = xv[row];
        v8h wv = *(const v8h*)(wp + row * DIM + c0);
        #pragma unroll
        for (int j = 0; j < 8; ++j) acc[j] += xi * (float)wv[j];
    }
    // lanes {l, l^8, l^16, l^24} hold the same columns -> reduce
    #pragma unroll
    for (int j = 0; j < 8; ++j) {
        acc[j] += __shfl_xor(acc[j], 8, 32);
        acc[j] += __shfl_xor(acc[j], 16, 32);
    }
    if (g == 0) {
        #pragma unroll
        for (int j = 0; j < 8; ++j)
            atomicAdd(&agg[d * DIM + c0 + j], acc[j]);
    }
}

// m = relu(agg/deg + conv_b)
__global__ void conv_post_kernel(const float* __restrict__ agg, const float* __restrict__ deg,
                                 const float* __restrict__ cb, float* __restrict__ m) {
    int t = blockIdx.x * blockDim.x + threadIdx.x;
    if (t >= NNODES * DIM) return;
    int n = t >> 6, j = t & 63;
    float dg = fmaxf(deg[n], 1.0f);
    m[t] = fmaxf(agg[t] / dg + cb[j], 0.0f);
}

// single-step GRU (torch gate order r,z,n)
__global__ void gru_kernel(const float* __restrict__ m, const float* __restrict__ h,
                           const float* __restrict__ wih, const float* __restrict__ whh,
                           const float* __restrict__ bih, const float* __restrict__ bhh,
                           float* __restrict__ hnew) {
    int t = blockIdx.x * blockDim.x + threadIdx.x;
    if (t >= NNODES * DIM) return;
    int n = t >> 6, j = t & 63;
    const float* mv = m + n * DIM;
    const float* hv = h + n * DIM;
    float ir = bih[j], iz = bih[64 + j], in_ = bih[128 + j];
    float hr = bhh[j], hz = bhh[64 + j], hn = bhh[128 + j];
    #pragma unroll 4
    for (int k = 0; k < DIM; ++k) {
        float mk = mv[k], hk = hv[k];
        ir += mk * wih[j * DIM + k];
        iz += mk * wih[(64 + j) * DIM + k];
        in_ += mk * wih[(128 + j) * DIM + k];
        hr += hk * whh[j * DIM + k];
        hz += hk * whh[(64 + j) * DIM + k];
        hn += hk * whh[(128 + j) * DIM + k];
    }
    float r = sigmoidf_(ir + hr);
    float z = sigmoidf_(iz + hz);
    float nn = tanhf(in_ + r * hn);
    hnew[t] = (1.0f - z) * nn + z * hv[j];
}

// Set2Set LSTM step (torch gate order i,f,g,o); ping-pong hh/cc
__global__ void lstm_kernel(const float* __restrict__ qstar, const float* __restrict__ hh,
                            const float* __restrict__ cc, const float* __restrict__ wih,
                            const float* __restrict__ whh, const float* __restrict__ bih,
                            const float* __restrict__ bhh, float* __restrict__ hhn,
                            float* __restrict__ ccn) {
    int t = blockIdx.x * blockDim.x + threadIdx.x;
    if (t >= NGRAPH * DIM) return;
    int b = t >> 6, j = t & 63;
    const float* q  = qstar + b * 2 * DIM;
    const float* hv = hh + b * DIM;
    float gi = bih[j] + bhh[j];
    float gf = bih[64 + j] + bhh[64 + j];
    float gg = bih[128 + j] + bhh[128 + j];
    float go = bih[192 + j] + bhh[192 + j];
    #pragma unroll 4
    for (int k = 0; k < 2 * DIM; ++k) {
        float qk = q[k];
        gi += qk * wih[j * 128 + k];
        gf += qk * wih[(64 + j) * 128 + k];
        gg += qk * wih[(128 + j) * 128 + k];
        go += qk * wih[(192 + j) * 128 + k];
    }
    #pragma unroll 4
    for (int k = 0; k < DIM; ++k) {
        float hk = hv[k];
        gi += hk * whh[j * 64 + k];
        gf += hk * whh[(64 + j) * 64 + k];
        gg += hk * whh[(128 + j) * 64 + k];
        go += hk * whh[(192 + j) * 64 + k];
    }
    float c = sigmoidf_(gf) * cc[t] + sigmoidf_(gi) * tanhf(gg);
    ccn[t] = c;
    hhn[t] = sigmoidf_(go) * tanhf(c);
}

// per-graph segment softmax attention + r_pool; writes q_star[b]
__global__ void attn_kernel(const float* __restrict__ feat, const int* __restrict__ batch,
                            const float* __restrict__ hh, float* __restrict__ ebuf,
                            float* __restrict__ qstar) {
    const int b = blockIdx.x;       // 64 graphs
    const int tid = threadIdx.x;    // 256 threads
    __shared__ float red[256];
    __shared__ float part[4][64];
    __shared__ float smax, ssum;
    const float* q = hh + b * DIM;

    // pass 1: e = <feat[n], q>, running max
    float lmax = -3.4e38f;
    for (int n = tid; n < NNODES; n += 256) {
        if (batch[n] == b) {
            const float* fv = feat + n * DIM;
            float e = 0.0f;
            #pragma unroll 8
            for (int k = 0; k < DIM; ++k) e += fv[k] * q[k];
            ebuf[n] = e;
            lmax = fmaxf(lmax, e);
        }
    }
    red[tid] = lmax; __syncthreads();
    for (int s = 128; s > 0; s >>= 1) {
        if (tid < s) red[tid] = fmaxf(red[tid], red[tid + s]);
        __syncthreads();
    }
    if (tid == 0) smax = red[0];
    __syncthreads();

    // pass 2: exp + sum
    float lsum = 0.0f;
    for (int n = tid; n < NNODES; n += 256) {
        if (batch[n] == b) {
            float ee = expf(ebuf[n] - smax);
            ebuf[n] = ee;
            lsum += ee;
        }
    }
    red[tid] = lsum; __syncthreads();
    for (int s = 128; s > 0; s >>= 1) {
        if (tid < s) red[tid] += red[tid + s];
        __syncthreads();
    }
    if (tid == 0) ssum = red[0];
    __syncthreads();

    // pass 3: r_pool[j] = sum ee * feat[:,j] / ssum
    int j = tid & 63, slot = tid >> 6;
    float acc = 0.0f;
    for (int n = slot; n < NNODES; n += 4)
        if (batch[n] == b) acc += ebuf[n] * feat[n * DIM + j];
    part[slot][j] = acc; __syncthreads();
    if (slot == 0) {
        float r = (part[0][j] + part[1][j] + part[2][j] + part[3][j]) / ssum;
        qstar[b * 128 + j] = q[j];
        qstar[b * 128 + 64 + j] = r;
    }
}

// d_out = concat(q_star.flat [8192], feat.flat [320000])
__global__ void finalize_kernel(const float* __restrict__ qstar, const float* __restrict__ feat,
                                float* __restrict__ out) {
    int t = blockIdx.x * blockDim.x + threadIdx.x;
    if (t < NGRAPH * 2 * DIM) out[t] = qstar[t];
    else if (t < NGRAPH * 2 * DIM + NNODES * DIM) out[t] = feat[t - NGRAPH * 2 * DIM];
}

extern "C" void kernel_launch(void* const* d_in, const int* in_sizes, int n_in,
                              void* d_out, int out_size, void* d_ws, size_t ws_size,
                              hipStream_t stream) {
    const float* x        = (const float*)d_in[0];
    const int*   eidx     = (const int*)d_in[1];
    const int*   batch    = (const int*)d_in[2];
    const float* eattr    = (const float*)d_in[3];
    const float* lin0_w   = (const float*)d_in[4];
    const float* lin0_b   = (const float*)d_in[5];
    const float* nn_w1    = (const float*)d_in[6];
    const float* nn_b1    = (const float*)d_in[7];
    const float* nn_w2    = (const float*)d_in[8];
    const float* nn_b2    = (const float*)d_in[9];
    const float* conv_b   = (const float*)d_in[10];
    const float* gru_wih  = (const float*)d_in[11];
    const float* gru_whh  = (const float*)d_in[12];
    const float* gru_bih  = (const float*)d_in[13];
    const float* gru_bhh  = (const float*)d_in[14];
    const float* lstm_wih = (const float*)d_in[15];
    const float* lstm_whh = (const float*)d_in[16];
    const float* lstm_bih = (const float*)d_in[17];
    const float* lstm_bhh = (const float*)d_in[18];
    float* out = (float*)d_out;
    const int* src = eidx;
    const int* dst = eidx + NEDGES;

    // workspace carve (256B aligned)
    char* w = (char*)d_ws;
    auto carve = [&](size_t bytes) -> void* {
        void* p = (void*)w;
        w += (bytes + 255) & ~(size_t)255;
        return p;
    };
    _Float16* hid  = (_Float16*)carve((size_t)NEDGES * HID * 2);      // 5.12 MB
    _Float16* w2t  = (_Float16*)carve((size_t)DD * HID * 2);          // 1.05 MB
    _Float16* ew   = (_Float16*)carve((size_t)NEDGES * DD * 2);       // 163.8 MB
    float* hA      = (float*)carve((size_t)NNODES * DIM * 4);
    float* hB      = (float*)carve((size_t)NNODES * DIM * 4);
    float* agg     = (float*)carve((size_t)NNODES * DIM * 4);
    float* mbuf    = (float*)carve((size_t)NNODES * DIM * 4);
    float* deg     = (float*)carve((size_t)NNODES * 4);
    float* ebuf    = (float*)carve((size_t)NNODES * 4);
    float* qstar   = (float*)carve((size_t)NGRAPH * 2 * DIM * 4);
    float* hh0     = (float*)carve((size_t)NGRAPH * DIM * 4);
    float* hh1     = (float*)carve((size_t)NGRAPH * DIM * 4);
    float* cc0     = (float*)carve((size_t)NGRAPH * DIM * 4);
    float* cc1     = (float*)carve((size_t)NGRAPH * DIM * 4);
    (void)in_sizes; (void)n_in; (void)out_size; (void)ws_size;

    const int BT = 256;
    // 1) lin0
    lin0_kernel<<<(NNODES * DIM + BT - 1) / BT, BT, 0, stream>>>(x, lin0_w, lin0_b, hA);
    // 2) edge MLP layer 1 (f16)
    edge_mlp1_kernel<<<(NEDGES * HID + BT - 1) / BT, BT, 0, stream>>>(eattr, nn_w1, nn_b1, hid);
    // 3) transpose+convert nn_w2
    w2_transpose_kernel<<<(DD * HID + BT - 1) / BT, BT, 0, stream>>>(nn_w2, w2t);
    // 4) big GEMM via WMMA: ew = hid @ nn_w2 + b2 ; 16x64 block per wave
    ew_gemm_wmma_kernel<<<(NEDGES / 16) * (DD / 64) / 8, BT, 0, stream>>>(hid, w2t, nn_b2, ew);
    // 5) degrees
    zero_f32<<<(NNODES + BT - 1) / BT, BT, 0, stream>>>(deg, NNODES);
    deg_kernel<<<(NEDGES + BT - 1) / BT, BT, 0, stream>>>(dst, deg);

    // 6) three NNConv + GRU iterations (ping-pong hA/hB)
    float* hcur = hA;
    float* hnxt = hB;
    for (int it = 0; it < 3; ++it) {
        zero_f32<<<(NNODES * DIM + BT - 1) / BT, BT, 0, stream>>>(agg, NNODES * DIM);
        msg_scatter_kernel<<<NEDGES / 8, BT, 0, stream>>>(hcur, ew, src, dst, agg);
        conv_post_kernel<<<(NNODES * DIM + BT - 1) / BT, BT, 0, stream>>>(agg, deg, conv_b, mbuf);
        gru_kernel<<<(NNODES * DIM + BT - 1) / BT, BT, 0, stream>>>(
            mbuf, hcur, gru_wih, gru_whh, gru_bih, gru_bhh, hnxt);
        float* tmp = hcur; hcur = hnxt; hnxt = tmp;
    }
    float* feat = hcur;

    // 7) Set2Set init
    zero_f32<<<(NGRAPH * 2 * DIM + BT - 1) / BT, BT, 0, stream>>>(qstar, NGRAPH * 2 * DIM);
    zero_f32<<<(NGRAPH * DIM + BT - 1) / BT, BT, 0, stream>>>(hh0, NGRAPH * DIM);
    zero_f32<<<(NGRAPH * DIM + BT - 1) / BT, BT, 0, stream>>>(cc0, NGRAPH * DIM);

    // 8) three Set2Set steps (ping-pong hh/cc)
    float* hhc = hh0; float* hhn = hh1;
    float* ccc = cc0; float* ccn = cc1;
    for (int s = 0; s < 3; ++s) {
        lstm_kernel<<<(NGRAPH * DIM + BT - 1) / BT, BT, 0, stream>>>(
            qstar, hhc, ccc, lstm_wih, lstm_whh, lstm_bih, lstm_bhh, hhn, ccn);
        attn_kernel<<<NGRAPH, BT, 0, stream>>>(feat, batch, hhn, ebuf, qstar);
        float* t1 = hhc; hhc = hhn; hhn = t1;
        float* t2 = ccc; ccc = ccn; ccn = t2;
    }

    // 9) output concat
    finalize_kernel<<<(NGRAPH * 2 * DIM + NNODES * DIM + BT - 1) / BT, BT, 0, stream>>>(
        qstar, feat, out);
}
